// Text_Encoder_63410897158619
// MI455X (gfx1250) — matrix operand order repaired and checked by reading, not compiled, not run on hardware
//
#include <hip/hip_runtime.h>
#include <hip/hip_bf16.h>

typedef __attribute__((ext_vector_type(16))) _Float16 v16h;
typedef __attribute__((ext_vector_type(8)))  _Float16 v8h;
typedef __attribute__((ext_vector_type(8)))  float    v8f;

#define B_  8192
#define T_  100
#define E_  128
#define H_  64
#define G4  256   // 4*H

__device__ __forceinline__ float sigm(float x) {
    return 1.0f / (1.0f + __expf(-x));
}
__device__ __forceinline__ float tanh_fast(float x) {
    x = fminf(fmaxf(x, -15.0f), 15.0f);
    float e = __expf(2.0f * x);
    return (e - 1.0f) / (e + 1.0f);
}

__global__ __launch_bounds__(256) void lstm_text_encoder(
    const int*   __restrict__ id_batch,  // [B]
    const int*   __restrict__ X_words,   // [D,T]
    const float* __restrict__ embed,     // [V,E]
    const float* __restrict__ hx0,       // [B,H]
    const float* __restrict__ cx0,       // [B,H]
    const float* __restrict__ W_ih,      // [4H,E]
    const float* __restrict__ W_hh,      // [4H,H]
    const float* __restrict__ b_ih,      // [4H]
    const float* __restrict__ b_hh,      // [4H]
    const float* __restrict__ W1,        // [32,H]
    const float* __restrict__ b1,        // [32]
    const float* __restrict__ W2,        // [5,32]
    const float* __restrict__ b2,        // [5]
    float*       __restrict__ out)       // [B,5]
{
    __shared__ __align__(16) _Float16 x_lds[16][E_];   // 4 KB  x_t tile (f16)
    __shared__ __align__(16) _Float16 h_lds[16][H_];   // 2 KB  h_t tile (f16)
    __shared__ __align__(16) float    g_lds[16][G4];   // 16 KB gate pre-activations
    __shared__ __align__(16) float    x1_lds[16][32];  // 2 KB  MLP hidden
    __shared__ int docs[16];

    const int tid  = threadIdx.x;
    const int lane = tid & 31;
    const int wave = tid >> 5;          // 0..7 (wave32)
    const int half = lane >> 4;         // 0 or 1
    const int l16  = lane & 15;
    const int b0   = blockIdx.x * 16;   // batch tile base

    if (tid < 16) docs[tid] = id_batch[b0 + tid];

    // Per-thread elementwise ownership: flat index p = tid + 256*q over 16x64.
    float c_reg[4], acc_reg[4];
#pragma unroll
    for (int q = 0; q < 4; ++q) {
        int p = tid + 256 * q;
        int m = p >> 6, j = p & 63;
        c_reg[q]   = cx0[(b0 + m) * H_ + j];
        acc_reg[q] = 0.0f;
        h_lds[m][j] = (_Float16)hx0[(b0 + m) * H_ + j];
    }

    // ---- Load weight fragments into registers (each wave owns 2 N-tiles) ----
    // B-fragment, contiguous k order chosen here (A uses the same): lane L holds N = L%16,
    // element i holds K = kbase + 16*(L/16) + i  -> 16 consecutive cols of W row.
    v16h wih[2][4];   // 4 K-steps over E=128
    v16h whh[2][2];   // 2 K-steps over H=64
    float bias[2];
#pragma unroll
    for (int tt = 0; tt < 2; ++tt) {
        const int nrow = (2 * wave + tt) * 16 + l16;   // row of W (gate output idx)
        bias[tt] = b_ih[nrow] + b_hh[nrow];
#pragma unroll
        for (int kk = 0; kk < 4; ++kk) {
            const float* src = W_ih + nrow * E_ + kk * 32 + 16 * half;
            v16h f;
#pragma unroll
            for (int i = 0; i < 16; ++i) f[i] = (_Float16)src[i];
            wih[tt][kk] = f;
        }
#pragma unroll
        for (int kk = 0; kk < 2; ++kk) {
            const float* src = W_hh + nrow * H_ + kk * 32 + 16 * half;
            v16h f;
#pragma unroll
            for (int i = 0; i < 16; ++i) f[i] = (_Float16)src[i];
            whh[tt][kk] = f;
        }
    }

    const int mrow = tid >> 4;          // staging row 0..15
    const int cseg = (tid & 15) * 8;    // staging col base (8 f32/thread)
    const int* wordrow = X_words + (long)docs[mrow] * T_;

    // ---- Software pipeline: prefetch embedding row segment for t=0 ----
    float xr[8];
    {
        const int word = wordrow[0];
        const float* er = embed + (long)word * E_ + cseg;
#pragma unroll
        for (int i = 0; i < 8; ++i) xr[i] = er[i];
    }

    // =========================== time loop ===========================
    for (int t = 0; t < T_; ++t) {
        __syncthreads();   // prev WMMA reads of x_lds / h_lds done

        // Commit prefetched x_t -> LDS (f16)
        {
            v8h hv;
#pragma unroll
            for (int i = 0; i < 8; ++i) hv[i] = (_Float16)xr[i];
            *(v8h*)(&x_lds[mrow][cseg]) = hv;
        }
        __syncthreads();

        // Issue the gather for t+1 now; it retires under this step's WMMAs
        if (t + 1 < T_) {
            const int word = wordrow[t + 1];
            const float* er = embed + (long)word * E_ + cseg;
#pragma unroll
            for (int i = 0; i < 8; ++i) xr[i] = er[i];
        }

        // g[16,256] = x_t @ W_ih^T + h @ W_hh^T + (b_ih+b_hh)
        v8f acc0, acc1;
#pragma unroll
        for (int r = 0; r < 8; ++r) { acc0[r] = bias[0]; acc1[r] = bias[1]; }

        // A-fragment, contiguous k order chosen here (matches B): lane L row M=L%16,
        // elem i -> K = kbase + 16*(L/16) + i
#pragma unroll
        for (int kk = 0; kk < 4; ++kk) {
            const int c0 = kk * 32 + 16 * half;
            v8h a_lo = *(const v8h*)(&x_lds[l16][c0]);
            v8h a_hi = *(const v8h*)(&x_lds[l16][c0 + 8]);
            v16h a;
#pragma unroll
            for (int i = 0; i < 8; ++i) { a[i] = a_lo[i]; a[8 + i] = a_hi[i]; }
            acc0 = __builtin_amdgcn_wmma_f32_16x16x32_f16(false, a, false, wih[0][kk],
                                                          (short)0, acc0, false, false);
            acc1 = __builtin_amdgcn_wmma_f32_16x16x32_f16(false, a, false, wih[1][kk],
                                                          (short)0, acc1, false, false);
        }
#pragma unroll
        for (int kk = 0; kk < 2; ++kk) {
            const int c0 = kk * 32 + 16 * half;
            v8h a_lo = *(const v8h*)(&h_lds[l16][c0]);
            v8h a_hi = *(const v8h*)(&h_lds[l16][c0 + 8]);
            v16h a;
#pragma unroll
            for (int i = 0; i < 8; ++i) { a[i] = a_lo[i]; a[8 + i] = a_hi[i]; }
            acc0 = __builtin_amdgcn_wmma_f32_16x16x32_f16(false, a, false, whh[0][kk],
                                                          (short)0, acc0, false, false);
            acc1 = __builtin_amdgcn_wmma_f32_16x16x32_f16(false, a, false, whh[1][kk],
                                                          (short)0, acc1, false, false);
        }

        // Spill g tiles: C/D layout -> element r is (M = r + 8*half, N = nbase + l16)
        {
            const int n0 = (2 * wave) * 16 + l16;
            const int n1 = n0 + 16;
#pragma unroll
            for (int r = 0; r < 8; ++r) {
                const int M = r + 8 * half;
                g_lds[M][n0] = acc0[r];
                g_lds[M][n1] = acc1[r];
            }
        }
        __syncthreads();

        // Elementwise LSTM cell update (c in regs, h -> LDS f16, acc in regs)
#pragma unroll
        for (int q = 0; q < 4; ++q) {
            const int p = tid + 256 * q;
            const int m = p >> 6, j = p & 63;
            const float ig = g_lds[m][j];
            const float fg = g_lds[m][H_ + j];
            const float gg = g_lds[m][2 * H_ + j];
            const float og = g_lds[m][3 * H_ + j];
            const float c  = sigm(fg) * c_reg[q] + sigm(ig) * tanh_fast(gg);
            c_reg[q] = c;
            const float h = sigm(og) * tanh_fast(c);
            acc_reg[q] += h;
            h_lds[m][j] = (_Float16)h;
        }
    }

    // =========================== MLP head ===========================
    __syncthreads();
#pragma unroll
    for (int q = 0; q < 4; ++q) {
        const int p = tid + 256 * q;
        const int m = p >> 6, j = p & 63;
        g_lds[m][j] = acc_reg[q] * (1.0f / (float)B_);   // reference divides by B
    }
    __syncthreads();

    for (int idx = tid; idx < 16 * 32; idx += 256) {
        const int m = idx >> 5, n1 = idx & 31;
        float s = b1[n1];
        const float* w = W1 + n1 * H_;
#pragma unroll 8
        for (int k = 0; k < H_; ++k) s += g_lds[m][k] * w[k];
        x1_lds[m][n1] = fmaxf(s, 0.0f);
    }
    __syncthreads();

    for (int idx = tid; idx < 16 * 5; idx += 256) {
        const int m = idx / 5, n2 = idx % 5;
        float s = b2[n2];
        const float* w = W2 + n2 * 32;
#pragma unroll
        for (int k = 0; k < 32; ++k) s += x1_lds[m][k] * w[k];
        out[(b0 + m) * 5 + n2] = sigm(s);
    }
}

extern "C" void kernel_launch(void* const* d_in, const int* in_sizes, int n_in,
                              void* d_out, int out_size, void* d_ws, size_t ws_size,
                              hipStream_t stream) {
    const int*   id_batch = (const int*)  d_in[0];
    const int*   X_words  = (const int*)  d_in[1];
    const float* embed    = (const float*)d_in[2];
    const float* hx0      = (const float*)d_in[3];
    const float* cx0      = (const float*)d_in[4];
    const float* W_ih     = (const float*)d_in[5];
    const float* W_hh     = (const float*)d_in[6];
    const float* b_ih     = (const float*)d_in[7];
    const float* b_hh     = (const float*)d_in[8];
    const float* W1       = (const float*)d_in[9];
    const float* b1       = (const float*)d_in[10];
    const float* W2       = (const float*)d_in[11];
    const float* b2       = (const float*)d_in[12];
    float* out = (float*)d_out;

    dim3 grid(B_ / 16);   // 512 tiles of 16 documents
    dim3 block(256);      // 8 wave32 waves
    hipLaunchKernelGGL(lstm_text_encoder, grid, block, 0, stream,
                       id_batch, X_words, embed, hx0, cx0,
                       W_ih, W_hh, b_ih, b_hh, W1, b1, W2, b2, out);
}
